// NAMTuring_81217831567554
// MI455X (gfx1250) — compile-verified, matrix operand order checked by reading, not run on hardware
//
#include <hip/hip_runtime.h>

// ---------------------------------------------------------------------------
// Types for CDNA5 WMMA / TDM
// ---------------------------------------------------------------------------
typedef __attribute__((ext_vector_type(16))) __bf16 v16bf;
typedef __attribute__((ext_vector_type(8)))  float  v8f;
typedef __attribute__((ext_vector_type(4)))  unsigned int u32x4;
typedef __attribute__((ext_vector_type(8)))  int          i32x8;
typedef __attribute__((ext_vector_type(4)))  int          i32x4;

union ABFrag {
    v16bf v;
    unsigned int u[8];
};

__device__ __forceinline__ unsigned short f2bf(float f) {
    unsigned int x = __float_as_uint(f);
    unsigned int r = x + 0x7fffu + ((x >> 16) & 1u);  // round-to-nearest-even
    return (unsigned short)(r >> 16);
}

__device__ __forceinline__ float sigm(float x) { return 1.0f / (1.0f + __expf(-x)); }

// ---------------------------------------------------------------------------
// Kernel 0: convert inputs + weights to bf16 in workspace
//   Abf : [8192 x 256]  from inputs
//   Wc  : [320 x 256]   rows 0..255 = W_val, rows 256..319 = W_ih
//   Wo  : [256 x 256]   W_out
// ---------------------------------------------------------------------------
__global__ void namt_convert(const float* __restrict__ inputs,
                             const float* __restrict__ W_val,
                             const float* __restrict__ W_ih,
                             const float* __restrict__ W_out,
                             unsigned short* __restrict__ Abf,
                             unsigned short* __restrict__ Wc,
                             unsigned short* __restrict__ Wo) {
    int i = blockIdx.x * blockDim.x + threadIdx.x;
    if (i < 8192 * 256) Abf[i] = f2bf(inputs[i]);
    if (i < 320 * 256) {
        int row = i >> 8;
        float v = (row < 256) ? W_val[i] : W_ih[(row - 256) * 256 + (i & 255)];
        Wc[i] = f2bf(v);
    }
    if (i < 256 * 256) Wo[i] = f2bf(W_out[i]);
}

// ---------------------------------------------------------------------------
// Shared WMMA tile: one wave computes a 16x16 f32 tile of A[MxK] @ W[NxK]^T
// (K = 256, bf16 inputs). Fragment packing per ISA 7.12.2.
// ---------------------------------------------------------------------------
__device__ __forceinline__ v8f wmma_tile_256(const unsigned short* __restrict__ A,
                                             const unsigned short* __restrict__ W,
                                             int m0, int n0) {
    const int lane = threadIdx.x & 31;
    const int half = lane >> 4;
    const int mr   = lane & 15;
    const int nc   = lane & 15;
    const unsigned short* arow = A + (size_t)(m0 + mr) * 256 + half * 8;
    const unsigned short* brow = W + (size_t)(n0 + nc) * 256 + half * 8;
    v8f acc = {};
#pragma unroll
    for (int kk = 0; kk < 256; kk += 32) {
        if (kk + 32 < 256) {
            __builtin_prefetch((const void*)(arow + kk + 32), 0, 0);
            __builtin_prefetch((const void*)(brow + kk + 32), 0, 0);
        }
        ABFrag a, b;
#pragma unroll
        for (int j = 0; j < 8; ++j) {
            int kbase = (j < 4 ? 2 * j : 16 + 2 * (j - 4)) + kk;
            a.u[j] = *(const unsigned int*)(arow + kbase);
            b.u[j] = *(const unsigned int*)(brow + kbase);
        }
        acc = __builtin_amdgcn_wmma_f32_16x16x32_bf16(
            /*neg_a=*/false, a.v, /*neg_b=*/false, b.v,
            /*c_mod=*/(short)0, acc, /*reuse_a=*/false, /*reuse_b=*/false);
    }
    return acc;
}

// ---------------------------------------------------------------------------
// Kernel 1: fused values/xg GEMM.  Out N = 320 (cols 0..255 -> values f32,
// cols 256..319 -> xg f32 with bias b_ih+b_hh).  grid(512, 20), block(32).
// ---------------------------------------------------------------------------
__global__ void namt_gemm_val_xg(const unsigned short* __restrict__ Abf,
                                 const unsigned short* __restrict__ Wc,
                                 const float* __restrict__ b_val,
                                 const float* __restrict__ b_ih,
                                 const float* __restrict__ b_hh,
                                 float* __restrict__ values,
                                 float* __restrict__ xg) {
    const int m0 = blockIdx.x * 16;
    const int n0 = blockIdx.y * 16;
    v8f acc = wmma_tile_256(Abf, Wc, m0, n0);
    const int lane = threadIdx.x & 31;
    const int half = lane >> 4;
    const int n = n0 + (lane & 15);
    const float bias = (n < 256) ? b_val[n] : (b_ih[n - 256] + b_hh[n - 256]);
#pragma unroll
    for (int r = 0; r < 8; ++r) {
        int m = m0 + r + 8 * half;
        float o = acc[r] + bias;
        if (n < 256) values[(size_t)m * 256 + n] = o;
        else         xg[(size_t)m * 64 + (n - 256)] = o;
    }
}

// ---------------------------------------------------------------------------
// Kernel 2: sequential LSTM + action heads.  One block, 256 threads.
// Emits per step: dirs_r [S,B,2,3], dirs_w [S,B,2,3], rw [S,B,2,2].
// ---------------------------------------------------------------------------
__global__ void namt_lstm(const float* __restrict__ xg,
                          const float* __restrict__ W_hh,
                          float* __restrict__ dr,
                          float* __restrict__ dw,
                          float* __restrict__ rwp) {
    __shared__ float h[32 * 16];
    __shared__ float cst[32 * 16];
    __shared__ float Whh[64 * 16];
    __shared__ float g[32 * 64];
    const int tid = threadIdx.x;
    for (int i = tid; i < 64 * 16; i += 256) Whh[i] = W_hh[i];
    for (int i = tid; i < 32 * 16; i += 256) { h[i] = 0.f; cst[i] = 0.f; }
    __syncthreads();
    for (int t = 0; t < 256; ++t) {
        // g = xg[t] + h @ W_hh^T   (32x64 outputs, K=16)
        for (int o = tid; o < 2048; o += 256) {
            int b = o >> 6, j = o & 63;
            float s = xg[(size_t)(t * 32 + b) * 64 + j];
#pragma unroll
            for (int u = 0; u < 16; ++u) s += h[b * 16 + u] * Whh[j * 16 + u];
            g[o] = s;
        }
        __syncthreads();
        // gates (torch order i,f,g,o)
        for (int o = tid; o < 512; o += 256) {
            int b = o >> 4, u = o & 15;
            float gi = g[b * 64 + u];
            float gf = g[b * 64 + 16 + u];
            float gg = g[b * 64 + 32 + u];
            float go = g[b * 64 + 48 + u];
            float cv = sigm(gf) * cst[o] + sigm(gi) * tanhf(gg);
            cst[o] = cv;
            h[o]   = sigm(go) * tanhf(cv);
        }
        __syncthreads();
        // action heads: h -> [B,2,8] -> softmax3, softmax3, sigmoid2
        if (tid < 64) {
            int b = tid >> 1, tp = tid & 1;
            const float* hh = &h[b * 16 + tp * 8];
            int base = (t * 32 + b) * 2 + tp;
            float m1 = fmaxf(hh[0], fmaxf(hh[1], hh[2]));
            float e0 = __expf(hh[0] - m1), e1 = __expf(hh[1] - m1), e2 = __expf(hh[2] - m1);
            float inv = 1.f / (e0 + e1 + e2);
            dr[base * 3 + 0] = e0 * inv; dr[base * 3 + 1] = e1 * inv; dr[base * 3 + 2] = e2 * inv;
            float m2 = fmaxf(hh[3], fmaxf(hh[4], hh[5]));
            float f0 = __expf(hh[3] - m2), f1 = __expf(hh[4] - m2), f2 = __expf(hh[5] - m2);
            float inv2 = 1.f / (f0 + f1 + f2);
            dw[base * 3 + 0] = f0 * inv2; dw[base * 3 + 1] = f1 * inv2; dw[base * 3 + 2] = f2 * inv2;
            rwp[base * 2 + 0] = sigm(hh[6]);
            rwp[base * 2 + 1] = sigm(hh[7]);
        }
        __syncthreads();
    }
}

// ---------------------------------------------------------------------------
// TDM helper: 1-D tile load (128 f32 = one values row) global -> LDS.
// Descriptor per CDNA5 ISA:
//   g0: [1:0]=count=1 | lds_addr[63:32] | global_addr[120:64] | type=2[127:126]
//   g1: data_size=2 (4B) [17:16], tensor_dim0=128 [79:48], tensor_dim1=1
//       [111:80], tile_dim0=128 [127:112], tile_dim1=1 [143:128],
//       tensor_dim0_stride=128 [207:160]
// Groups 2/3 zero (<=2D tensor).  Issued by wave 0 only; TDM ignores EXEC.
// ---------------------------------------------------------------------------
__device__ __forceinline__ void tdm_load_row128(const float* gsrc, float* lds_dst) {
#if __has_builtin(__builtin_amdgcn_tensor_load_to_lds)
    unsigned long long ga = (unsigned long long)(uintptr_t)gsrc;
    unsigned ldsa = (unsigned)(uintptr_t)lds_dst;   // low 32 bits = LDS offset
    u32x4 g0;
    g0[0] = 1u;                                     // count=1, user-mode
    g0[1] = ldsa;                                   // lds_addr
    g0[2] = (unsigned)(ga & 0xffffffffu);           // global_addr[31:0]
    g0[3] = (unsigned)(ga >> 32) | (2u << 30);      // global_addr[56:32] | type=2
    i32x8 g1;
    g1[0] = 0x00020000;                             // data_size = 4 bytes
    g1[1] = (int)(128u << 16);                      // tensor_dim0 = 128 (lo16)
    g1[2] = (int)(1u << 16);                        // dim0 hi=0 | tensor_dim1 = 1
    g1[3] = (int)(128u << 16);                      // dim1 hi=0 | tile_dim0 = 128
    g1[4] = 1;                                      // tile_dim1 = 1, tile_dim2 = 0
    g1[5] = 128;                                    // tensor_dim0_stride = 128
    g1[6] = 0;
    g1[7] = 0;
    i32x4 z4 = {};
#if defined(__clang_major__) && (__clang_major__ >= 23)
    i32x8 z8 = {};
    __builtin_amdgcn_tensor_load_to_lds(g0, g1, z4, z4, z8, 0);
#else
    __builtin_amdgcn_tensor_load_to_lds(g0, g1, z4, z4, 0);
#endif
#else
    // Fallback: plain cooperative copy by the issuing wave (32 lanes x 4 rows)
    int l = threadIdx.x & 31;
    for (int j = 0; j < 4; ++j) lds_dst[l + 32 * j] = gsrc[l + 32 * j];
#endif
}

__device__ __forceinline__ void tdm_wait0() {
#if __has_builtin(__builtin_amdgcn_s_wait_tensorcnt)
    __builtin_amdgcn_s_wait_tensorcnt(0);
#endif
}

// ---------------------------------------------------------------------------
// Kernel 3: soft-tape scan.  One block per (batch, tape) head (64 blocks,
// 256 threads).  Entire 256x128 f32 tape lives in LDS (128 KB of the WGP's
// 320 KB) for all 256 sequential steps.  Per-step values row is streamed
// into a double-buffered LDS slot by the Tensor Data Mover, overlapped with
// the oldval/tape-update reductions.  Dynamic LDS = 135,680 bytes.
// ---------------------------------------------------------------------------
__global__ void namt_tape_scan(const float* __restrict__ values,
                               const float* __restrict__ dr,
                               const float* __restrict__ dw,
                               const float* __restrict__ rwg,
                               unsigned short* __restrict__ reads_bf,
                               float* __restrict__ out_tape,
                               float* __restrict__ out_rpos,
                               float* __restrict__ out_wpos) {
    extern __shared__ float sm[];
    float* tape = sm;                  // 256*128
    float* rpos = tape + 256 * 128;    // 256
    float* wpos = rpos + 256;          // 256
    float* vbuf = wpos + 256;          // 2 x 128 (double buffer)
    float* oldv = vbuf + 256;          // 128
    float* part = oldv + 128;          // 256

    const int tid = threadIdx.x;
    const int b  = blockIdx.x >> 1;
    const int tp = blockIdx.x & 1;
    for (int i = tid; i < 256 * 128; i += 256) tape[i] = 0.f;
    rpos[tid] = (tid == 0) ? 1.f : 0.f;
    wpos[tid] = rpos[tid];
    __syncthreads();

    const int c  = tid & 127;
    const int lh = tid >> 7;           // 0/1: which half of the L reduction

    // prologue: TDM-load values row for t=0 into buffer 0
    if (tid < 32) {
        tdm_load_row128(values + (size_t)((0 * 32 + b) * 2 + tp) * 128, vbuf);
        tdm_wait0();
    }
    __syncthreads();

    for (int t = 0; t < 256; ++t) {
        const int head = (t * 32 + b) * 2 + tp;
        const float rd0 = dr[head * 3 + 0], rd1 = dr[head * 3 + 1], rd2 = dr[head * 3 + 2];
        const float wd0 = dw[head * 3 + 0], wd1 = dw[head * 3 + 1], wd2 = dw[head * 3 + 2];
        const float rw0 = rwg[head * 2 + 0], rw1 = rwg[head * 2 + 1];
        const float* vb = vbuf + (t & 1) * 128;

        // kick off async TDM fetch of next step's row into the other buffer;
        // it overlaps the two 256-deep LDS reductions below.
        if (t < 255 && tid < 32)
            tdm_load_row128(values + (size_t)(head + 64) * 128,
                            vbuf + ((t + 1) & 1) * 128);

        // oldval[c] = sum_l tape[l][c] * wpos[l]   (split reduction over l)
        float s = 0.f;
        for (int l = lh * 128; l < lh * 128 + 128; ++l) s += tape[l * 128 + c] * wpos[l];
        part[tid] = s;
        __syncthreads();
        if (tid < 128) oldv[tid] = part[tid] + part[128 + tid];
        __syncthreads();
        // tape += wpos (x) (v - oldval) * rw1
        for (int idx = tid; idx < 256 * 128; idx += 256) {
            int l = idx >> 7, cc = idx & 127;
            tape[idx] += wpos[l] * (vb[cc] - oldv[cc]) * rw1;
        }
        __syncthreads();
        // read_out[c] = rw0 * sum_l tape[l][c] * rpos[l]  (updated tape, old rpos)
        float s2 = 0.f;
        for (int l = lh * 128; l < lh * 128 + 128; ++l) s2 += tape[l * 128 + c] * rpos[l];
        part[tid] = s2;
        __syncthreads();
        if (tid < 128) {
            float ro = (part[tid] + part[128 + tid]) * rw0;
            reads_bf[(size_t)(t * 32 + b) * 256 + tp * 128 + tid] = f2bf(ro);
        }
        // position updates (both built from OLD rpos — faithful to source)
        float rp_p = rpos[(tid + 255) & 255];  // rpos[l-1]
        float rp_c = rpos[tid];
        float rp_n = rpos[(tid + 1) & 255];    // rpos[l+1]
        float wp_c = wpos[tid];
        __syncthreads();
        wpos[tid] = rp_n * wd0 + wp_c * wd1 + rp_p * wd2;
        rpos[tid] = rp_n * rd0 + rp_c * rd1 + rp_p * rd2;
        // make sure next step's TDM row has landed before anyone reads it
        tdm_wait0();
        __syncthreads();
    }

    // final state out: tape [L,B,2,128], rpos/wpos [L,B,2]
    for (int idx = tid; idx < 256 * 128; idx += 256) {
        int l = idx >> 7, cc = idx & 127;
        out_tape[(size_t)((l * 32 + b) * 2 + tp) * 128 + cc] = tape[idx];
    }
    out_rpos[(tid * 32 + b) * 2 + tp] = rpos[tid];
    out_wpos[(tid * 32 + b) * 2 + tp] = wpos[tid];
}

// ---------------------------------------------------------------------------
// Kernel 4: outputs = reads @ W_out^T + b_out.  grid(512, 16), block(32).
// ---------------------------------------------------------------------------
__global__ void namt_gemm_out(const unsigned short* __restrict__ Rbf,
                              const unsigned short* __restrict__ Wo,
                              const float* __restrict__ b_out,
                              float* __restrict__ outputs) {
    const int m0 = blockIdx.x * 16;
    const int n0 = blockIdx.y * 16;
    v8f acc = wmma_tile_256(Rbf, Wo, m0, n0);
    const int lane = threadIdx.x & 31;
    const int half = lane >> 4;
    const int n = n0 + (lane & 15);
    const float bias = b_out[n];
#pragma unroll
    for (int r = 0; r < 8; ++r) {
        int m = m0 + r + 8 * half;
        outputs[(size_t)m * 256 + n] = acc[r] + bias;
    }
}

// ---------------------------------------------------------------------------
// Host-side launcher
// ---------------------------------------------------------------------------
extern "C" void kernel_launch(void* const* d_in, const int* in_sizes, int n_in,
                              void* d_out, int out_size, void* d_ws, size_t ws_size,
                              hipStream_t stream) {
    (void)in_sizes; (void)n_in; (void)out_size; (void)ws_size;
    const float* inputs = (const float*)d_in[0];
    // d_in[1] = tapelen (== 256, compiled in)
    const float* W_ih  = (const float*)d_in[2];
    const float* W_hh  = (const float*)d_in[3];
    const float* b_ih  = (const float*)d_in[4];
    const float* b_hh  = (const float*)d_in[5];
    const float* W_val = (const float*)d_in[6];
    const float* b_val = (const float*)d_in[7];
    const float* W_out = (const float*)d_in[8];
    const float* b_out = (const float*)d_in[9];

    char* ws = (char*)d_ws;
    unsigned short* Abf      = (unsigned short*)(ws + 0);          // 8192*256 bf16
    float*          values   = (float*)(ws + 4194304);             // 8192*256 f32
    float*          xg       = (float*)(ws + 12582912);            // 8192*64  f32
    unsigned short* Wc       = (unsigned short*)(ws + 14680064);   // 320*256  bf16
    unsigned short* Wo       = (unsigned short*)(ws + 14843904);   // 256*256  bf16
    float*          dr       = (float*)(ws + 14974976);            // 256*32*2*3
    float*          dwv      = (float*)(ws + 15171584);            // 256*32*2*3
    float*          rwp      = (float*)(ws + 15368192);            // 256*32*2*2
    unsigned short* reads_bf = (unsigned short*)(ws + 15499264);   // 8192*256 bf16

    float* out_outputs = (float*)d_out;                 // [256,32,256]
    float* out_tape    = out_outputs + 2097152;         // [256,32,2,128]
    float* out_rpos    = out_tape + 2097152;            // [256,32,2]
    float* out_wpos    = out_rpos + 16384;              // [256,32,2]

    namt_convert<<<8192, 256, 0, stream>>>(inputs, W_val, W_ih, W_out, Abf, Wc, Wo);
    namt_gemm_val_xg<<<dim3(512, 20), 32, 0, stream>>>(Abf, Wc, b_val, b_ih, b_hh,
                                                       values, xg);
    namt_lstm<<<1, 256, 0, stream>>>(xg, W_hh, dr, dwv, rwp);
    namt_tape_scan<<<64, 256, 135680, stream>>>(values, dr, dwv, rwp, reads_bf,
                                                out_tape, out_rpos, out_wpos);
    namt_gemm_out<<<dim3(512, 16), 32, 0, stream>>>(reads_bf, Wo, b_out, out_outputs);
}
